// Cross_Attention_38439957299379
// MI455X (gfx1250) — compile-verified, matrix operand order checked by reading, not run on hardware
//
#include <hip/hip_runtime.h>
#include <hip/hip_bf16.h>
#include <stdint.h>

// ---------------------------------------------------------------------------
// CDNA5 (gfx1250) fused cross-attention:  q=x@Wq (+RoPE) -> attn -> @Wo
// Matrix math: v_wmma_f32_16x16x32_bf16 (wave32, f32 accumulate).
// K/V^T staging in attention: Tensor Data Mover (tensor_load_to_lds) with
// descriptor-side LDS padding; GEMM uses double-buffered LDS tiles.
// ---------------------------------------------------------------------------

typedef __bf16 bf16;
typedef __attribute__((ext_vector_type(16))) __bf16 v16bf;
typedef __attribute__((ext_vector_type(8)))  __bf16 v8bf;
typedef __attribute__((ext_vector_type(4)))  __bf16 v4bf;
typedef __attribute__((ext_vector_type(8)))  float  v8f;
typedef __attribute__((ext_vector_type(4)))  unsigned int u32x4;
typedef __attribute__((ext_vector_type(8)))  int i32x8;
typedef __attribute__((ext_vector_type(4)))  int i32x4;

#define DIM        4096
#define N_HEADS    32
#define HEAD_DIM   128
#define N_KV_HEADS 8
#define REPEATS    4
#define NUM_SEQS   4
#define Q_LEN      2048
#define KV_LEN     512
#define S_TOT      (NUM_SEQS * Q_LEN)    // 8192
#define SK_TOT     (NUM_SEQS * KV_LEN)   // 2048
#define ATTN_SCALE 0.08838834764831845f  // 128^-0.5
#define NEG_LN_THETA_OVER_64 (-0.14391156832f) // -ln(10000)/64

// LDS row strides (elements), padded to break 64-bank conflicts.
// Pads are 4 DWORDs (8 bf16) so the TDM descriptor pad fields can recreate them.
#define KLD 136   // K tile rows: 128 + 8
#define VLD 520   // V^T rows:    512 + 8
#define PLD 40    // P scratch:   32 + 8
#define K_ELE (KV_LEN * KLD)
#define V_ELE (HEAD_DIM * VLD)
#define P_ELE (8 * 16 * PLD)

#define HAVE_TDM (__has_builtin(__builtin_amdgcn_tensor_load_to_lds) && \
                  __has_builtin(__builtin_amdgcn_s_wait_tensorcnt))

static __device__ __forceinline__ v8f wmma_bf16(v16bf a, v16bf b, v8f c) {
  // D = A(16x32) * B(32x16) + C, f32 accumulate
  return __builtin_amdgcn_wmma_f32_16x16x32_bf16(false, a, false, b,
                                                 (short)0, c, false, false);
}

// 16-bit A/B fragment: per-lane two 16B loads at K-offsets base and base+16,
// where base = (lane<16 ? 0 : 8). Caller passes p = rowBase + base.
static __device__ __forceinline__ v16bf load_frag(const bf16* p) {
  v8bf lo = *(const v8bf*)(p);
  v8bf hi = *(const v8bf*)(p + 16);
  v16bf r;
#pragma unroll
  for (int i = 0; i < 8; ++i) { r[i] = lo[i]; r[i + 8] = hi[i]; }
  return r;
}

// ------------------------------- prep kernels ------------------------------

__global__ void k_cvt_bf16(const float* __restrict__ in, bf16* __restrict__ out,
                           int n4) {
  int i = blockIdx.x * blockDim.x + threadIdx.x;
  if (i >= n4) return;
  float4 v = ((const float4*)in)[i];
  v4bf o;
  o[0] = (bf16)v.x; o[1] = (bf16)v.y; o[2] = (bf16)v.z; o[3] = (bf16)v.w;
  *(v4bf*)(out + (size_t)i * 4) = o;
}

// W[K][N] f32 -> WT[N][K] bf16 via 32x32 LDS tile
__global__ void k_transcvt(const float* __restrict__ W, bf16* __restrict__ WT,
                           int K, int N) {
  __shared__ float tile[32][33];
  int k0 = blockIdx.y * 32, n0 = blockIdx.x * 32;
  for (int r = threadIdx.y; r < 32; r += 8)
    tile[r][threadIdx.x] = W[(size_t)(k0 + r) * N + n0 + threadIdx.x];
  __syncthreads();
  for (int r = threadIdx.y; r < 32; r += 8)
    WT[(size_t)(n0 + r) * K + k0 + threadIdx.x] = (bf16)tile[threadIdx.x][r];
}

// xk[b*512+kv][kvh*128+d] f32 -> RoPE -> kout[(b*8+kvh)*512+kv][d] bf16
__global__ void k_rope_k(const float* __restrict__ xk, bf16* __restrict__ kout) {
  int idx = blockIdx.x * 256 + threadIdx.x;     // one thread per (row, pair)
  int row = idx >> 9;                            // [0, 2048)
  int col = (idx & 511) * 2;                     // [0, 1024)
  int kv = row & (KV_LEN - 1);
  int b  = row >> 9;
  int kvh = col >> 7;
  int d   = col & 127;
  int j   = d >> 1;
  float tr = xk[(size_t)row * (N_KV_HEADS * HEAD_DIM) + col];
  float ti = xk[(size_t)row * (N_KV_HEADS * HEAD_DIM) + col + 1];
  float ang = (float)kv * __expf(NEG_LN_THETA_OVER_64 * (float)j);
  float c, s;
  __sincosf(ang, &s, &c);
  size_t o = ((size_t)((b * N_KV_HEADS + kvh) * KV_LEN + kv)) * HEAD_DIM + d;
  kout[o]     = (bf16)(tr * c - ti * s);
  kout[o + 1] = (bf16)(tr * s + ti * c);
}

// xv[b*512+kv][kvh*128+d] f32 -> vt[(b*8+kvh)*128+d][kv] bf16 (transposed)
__global__ void k_vtrans(const float* __restrict__ xv, bf16* __restrict__ vt) {
  int idx = blockIdx.x * 256 + threadIdx.x;
  int kv  = idx & (KV_LEN - 1);
  int d   = (idx >> 9) & (HEAD_DIM - 1);
  int kvh = (idx >> 16) & (N_KV_HEADS - 1);
  int b   = idx >> 19;
  vt[((size_t)((b * N_KV_HEADS + kvh) * HEAD_DIM + d)) * KV_LEN + kv] =
      (bf16)xv[(size_t)(b * KV_LEN + kv) * (N_KV_HEADS * HEAD_DIM)
               + kvh * HEAD_DIM + d];
}

// ----------------------------- bf16 WMMA GEMM ------------------------------
// C[M][N] = A[M][K] * BT[N][K]^T.  128x128 tile, 8 waves, 2x4 wmma tiles/wave.
// Double-buffered LDS: one barrier per K-step; next tile's global loads are in
// flight (registers) while WMMAs consume the current tile.
// mode 0: f32 -> Cf.   mode 1: fused RoPE -> bf16 Cb (pos = row & (Q_LEN-1)).
__global__ __launch_bounds__(256) void k_gemm_nt(
    const bf16* __restrict__ A, const bf16* __restrict__ BT,
    float* __restrict__ Cf, bf16* __restrict__ Cb,
    int M, int N, int K, int mode) {
  __shared__ bf16 As[2][128 * 40];
  __shared__ bf16 Bs[2][128 * 40];
  const int tid = threadIdx.x;
  const int lane = tid & 31;
  const int w = tid >> 5;
  const int bm = blockIdx.y * 128;
  const int bn = blockIdx.x * 128;
  const int waveM = (w & 3) * 32;
  const int waveN = (w >> 2) * 64;
  const int base8 = (lane < 16) ? 0 : 8;
  const int l16 = lane & 15;

  v8f acc[2][4];
#pragma unroll
  for (int a = 0; a < 2; ++a)
#pragma unroll
    for (int b = 0; b < 4; ++b)
#pragma unroll
      for (int i = 0; i < 8; ++i) acc[a][b][i] = 0.0f;

  const int r = tid >> 1;          // 2 threads per tile row
  const int seg = (tid & 1) * 16;  // 16 bf16 (2x uint4) each
  const bf16* aRow = A + (size_t)(bm + r) * K + seg;
  const bf16* bRow = BT + (size_t)(bn + r) * K + seg;

  // prologue: fill buffer 0
  uint4 a0 = ((const uint4*)aRow)[0], a1 = ((const uint4*)aRow)[1];
  uint4 b0 = ((const uint4*)bRow)[0], b1 = ((const uint4*)bRow)[1];
  {
    uint4* ad = (uint4*)(As[0] + r * 40 + seg);
    uint4* bd = (uint4*)(Bs[0] + r * 40 + seg);
    ad[0] = a0; ad[1] = a1; bd[0] = b0; bd[1] = b1;
  }
  __syncthreads();

  int cur = 0;
  for (int k0 = 0; k0 < K; k0 += 32) {
    const bool hasNext = (k0 + 32) < K;
    if (hasNext) {  // start next tile's global loads immediately
      const uint4* ag = (const uint4*)(aRow + k0 + 32);
      const uint4* bg = (const uint4*)(bRow + k0 + 32);
      a0 = ag[0]; a1 = ag[1]; b0 = bg[0]; b1 = bg[1];
      __builtin_prefetch(aRow + k0 + 64, 0, 1);
      __builtin_prefetch(bRow + k0 + 64, 0, 1);
    }
    const bf16* Ac = As[cur];
    const bf16* Bc = Bs[cur];
    v16bf af[2], bfr[4];
#pragma unroll
    for (int tr = 0; tr < 2; ++tr)
      af[tr] = load_frag(Ac + (waveM + tr * 16 + l16) * 40 + base8);
#pragma unroll
    for (int tc = 0; tc < 4; ++tc)
      bfr[tc] = load_frag(Bc + (waveN + tc * 16 + l16) * 40 + base8);
#pragma unroll
    for (int tr = 0; tr < 2; ++tr)
#pragma unroll
      for (int tc = 0; tc < 4; ++tc)
        acc[tr][tc] = wmma_bf16(af[tr], bfr[tc], acc[tr][tc]);
    if (hasNext) {
      uint4* ad = (uint4*)(As[cur ^ 1] + r * 40 + seg);
      uint4* bd = (uint4*)(Bs[cur ^ 1] + r * 40 + seg);
      ad[0] = a0; ad[1] = a1; bd[0] = b0; bd[1] = b1;
      __syncthreads();   // publishes next buffer; own DS reads already waited
      cur ^= 1;
    }
  }

  const int halfr = (lane >> 4) << 3;   // C layout: M = halfr + vgpr index
  if (mode == 0) {
#pragma unroll
    for (int tr = 0; tr < 2; ++tr)
#pragma unroll
      for (int tc = 0; tc < 4; ++tc)
#pragma unroll
        for (int i = 0; i < 8; ++i) {
          int row = bm + waveM + tr * 16 + halfr + i;
          int col = bn + waveN + tc * 16 + l16;
          Cf[(size_t)row * N + col] = acc[tr][tc][i];
        }
  } else {
    // RoPE epilogue: even/odd column pair sits in adjacent lanes -> shfl_xor(1)
#pragma unroll
    for (int tr = 0; tr < 2; ++tr)
#pragma unroll
      for (int tc = 0; tc < 4; ++tc) {
        int col = bn + waveN + tc * 16 + l16;
        int j = (col & (HEAD_DIM - 1)) >> 1;
        float invf = __expf(NEG_LN_THETA_OVER_64 * (float)j);
#pragma unroll
        for (int i = 0; i < 8; ++i) {
          int row = bm + waveM + tr * 16 + halfr + i;
          float v = acc[tr][tc][i];
          float o = __shfl_xor(v, 1, 32);
          float ang = (float)(row & (Q_LEN - 1)) * invf;
          float sn, cs;
          __sincosf(ang, &sn, &cs);
          float res = ((lane & 1) == 0) ? (v * cs - o * sn)   // real slot
                                        : (o * sn + v * cs);  // imag slot
          Cb[(size_t)row * N + col] = (bf16)res;
        }
      }
  }
}

// ----------------------------- attention kernel ----------------------------
// Block = 256 thr (8 waves). Block handles 128 q rows of one (seq, head).
// K[512][128] and V^T[128][512] pulled into LDS by the Tensor Data Mover
// (descriptor-side padding recreates the bank-conflict strides), ~276 KB of
// the 320 KB WGP LDS. Each wave: 16 q rows, online softmax over KV steps of 32.
__global__ __launch_bounds__(256) void k_attn(
    const bf16* __restrict__ qg, const bf16* __restrict__ kg,
    const bf16* __restrict__ vtg, bf16* __restrict__ outg) {
  extern __shared__ __align__(16) char smem[];
  bf16* Ks = (bf16*)smem;          // [512][KLD]
  bf16* Vs = Ks + K_ELE;           // [128][VLD]  (V transposed: rows are d)
  bf16* Ps = Vs + V_ELE;           // per-wave [16][PLD] scratch

  const int tid = threadIdx.x;
  const int lane = tid & 31;
  const int w = tid >> 5;
  const int l16 = lane & 15;
  const int base8 = (lane < 16) ? 0 : 8;
  const int halfr = (lane >> 4) << 3;

  const int qtile = blockIdx.x;        // 0..15
  const int bh = blockIdx.y;           // b*32 + h
  const int b = bh >> 5;
  const int h = bh & 31;
  const int kvh = h >> 2;              // grouped: kv head = h / REPEATS

  const bf16* ksrc = kg + (size_t)(b * N_KV_HEADS + kvh) * KV_LEN * HEAD_DIM;
  const bf16* vsrc = vtg + (size_t)(b * N_KV_HEADS + kvh) * HEAD_DIM * KV_LEN;

#if HAVE_TDM
  // ---- TDM: two 2-D tile descriptors, issued by wave 0 only (wave-level DMA,
  // EXEC-ignored; TENSORcnt is per-wave so the issuing wave waits, then the
  // workgroup barrier releases everyone else).
  if (w == 0) {
    unsigned long long ka = (unsigned long long)ksrc;
    u32x4 g0;
    g0[0] = 1u;                                         // count=1, user mode
    g0[1] = 0u;                                         // lds_addr: Ks @ 0
    g0[2] = (unsigned)(ka & 0xffffffffu);               // global_addr lo
    g0[3] = (unsigned)((ka >> 32) & 0x01ffffffu) | (2u << 30);  // hi | type=2
    i32x8 g1;
    // data_size=2B(1), pad_enable, pad_interval=64 DW (code 5), pad_amount=4 DW (code 3)
    g1[0] = (1 << 16) | (1 << 20) | (5 << 22) | (3 << 25);
    g1[1] = (HEAD_DIM & 0xffff) << 16;   // tensor_dim0 = 128   (bits 79:48)
    g1[2] = (KV_LEN & 0xffff) << 16;     // tensor_dim1 = 512   (bits 111:80)
    g1[3] = (HEAD_DIM << 16);            // tile_dim0   = 128   (bits 127:112)
    g1[4] = KV_LEN;                      // tile_dim1   = 512, tile_dim2 = 0
    g1[5] = HEAD_DIM;                    // tensor_dim0_stride = 128
    g1[6] = 0;
    g1[7] = 0;
    i32x4 z4; z4[0] = z4[1] = z4[2] = z4[3] = 0;         // 2-D: groups 2/3 off
    i32x8 z8;
#pragma unroll
    for (int i = 0; i < 8; ++i) z8[i] = 0;
    __builtin_amdgcn_tensor_load_to_lds(g0, g1, z4, z4, z8, 0);

    unsigned long long va = (unsigned long long)vsrc;
    g0[1] = (unsigned)(K_ELE * sizeof(bf16));            // lds_addr: Vs
    g0[2] = (unsigned)(va & 0xffffffffu);
    g0[3] = (unsigned)((va >> 32) & 0x01ffffffu) | (2u << 30);
    // pad_interval=256 DW (code 7) for 512-bf16 rows, pad_amount=4 DW
    g1[0] = (1 << 16) | (1 << 20) | (7 << 22) | (3 << 25);
    g1[1] = (KV_LEN & 0xffff) << 16;     // tensor_dim0 = 512
    g1[2] = (HEAD_DIM & 0xffff) << 16;   // tensor_dim1 = 128
    g1[3] = (KV_LEN << 16);              // tile_dim0   = 512
    g1[4] = HEAD_DIM;                    // tile_dim1   = 128
    g1[5] = KV_LEN;                      // tensor_dim0_stride = 512
    g1[6] = 0;
    g1[7] = 0;
    __builtin_amdgcn_tensor_load_to_lds(g0, g1, z4, z4, z8, 0);
    __builtin_amdgcn_s_wait_tensorcnt(0);
  }
  __syncthreads();
#else
  // ---- fallback: cooperative vector copy ----
  for (int rr = tid; rr < KV_LEN; rr += 256) {
    const uint4* s = (const uint4*)(ksrc + (size_t)rr * HEAD_DIM);
    uint4* d = (uint4*)(Ks + rr * KLD);
#pragma unroll
    for (int c = 0; c < 16; ++c) d[c] = s[c];
  }
  {
    int vr = tid >> 1;
    int vh = (tid & 1) * 32;
    const uint4* s = (const uint4*)(vsrc + (size_t)vr * KV_LEN) + vh;
    uint4* d = (uint4*)(Vs + vr * VLD) + vh;
#pragma unroll
    for (int c = 0; c < 32; ++c) d[c] = s[c];
  }
  __syncthreads();
#endif

  // ---- Q fragments for this wave's 16 rows (kept in registers) ----
  const int srow = b * Q_LEN + qtile * 128 + w * 16;  // global q row base
  v16bf qf[4];
#pragma unroll
  for (int dstep = 0; dstep < 4; ++dstep)
    qf[dstep] = load_frag(qg + (size_t)(srow + l16) * (N_HEADS * HEAD_DIM)
                          + h * HEAD_DIM + dstep * 32 + base8);

  v8f acc[8];
#pragma unroll
  for (int t = 0; t < 8; ++t)
#pragma unroll
    for (int i = 0; i < 8; ++i) acc[t][i] = 0.0f;
  float mrow[8], lrow[8], fac[8];
#pragma unroll
  for (int i = 0; i < 8; ++i) { mrow[i] = -1e30f; lrow[i] = 0.0f; }

  bf16* Pw = Ps + w * 16 * PLD;

  for (int kv0 = 0; kv0 < KV_LEN; kv0 += 32) {
    // S tiles: s0 = Q*K^T for kv cols [kv0,kv0+16), s1 for [kv0+16,kv0+32)
    v8f s0, s1;
#pragma unroll
    for (int i = 0; i < 8; ++i) { s0[i] = 0.0f; s1[i] = 0.0f; }
#pragma unroll
    for (int dstep = 0; dstep < 4; ++dstep) {
      v16bf kf0 = load_frag(Ks + (kv0 + l16) * KLD + dstep * 32 + base8);
      v16bf kf1 = load_frag(Ks + (kv0 + 16 + l16) * KLD + dstep * 32 + base8);
      s0 = wmma_bf16(qf[dstep], kf0, s0);
      s1 = wmma_bf16(qf[dstep], kf1, s1);
    }
    // online softmax per row (row reductions across 16 lanes of each half)
#pragma unroll
    for (int i = 0; i < 8; ++i) {
      float a = s0[i] * ATTN_SCALE;
      float bb = s1[i] * ATTN_SCALE;
      float mx = fmaxf(a, bb);
      mx = fmaxf(mx, __shfl_xor(mx, 1, 32));
      mx = fmaxf(mx, __shfl_xor(mx, 2, 32));
      mx = fmaxf(mx, __shfl_xor(mx, 4, 32));
      mx = fmaxf(mx, __shfl_xor(mx, 8, 32));
      float mnew = fmaxf(mrow[i], mx);
      float f = __expf(mrow[i] - mnew);
      float p0 = __expf(a - mnew);
      float p1 = __expf(bb - mnew);
      float sum = p0 + p1;
      sum += __shfl_xor(sum, 1, 32);
      sum += __shfl_xor(sum, 2, 32);
      sum += __shfl_xor(sum, 4, 32);
      sum += __shfl_xor(sum, 8, 32);
      lrow[i] = lrow[i] * f + sum;
      mrow[i] = mnew;
      fac[i] = f;
      int prow = halfr + i;  // C-layout row -> P row
      Pw[prow * PLD + l16]      = (bf16)p0;
      Pw[prow * PLD + l16 + 16] = (bf16)p1;
    }
    // rescale running O
#pragma unroll
    for (int t = 0; t < 8; ++t)
#pragma unroll
      for (int i = 0; i < 8; ++i) acc[t][i] *= fac[i];
    // P back as A-fragment (wave-private LDS; per-wave DS ops are in-order)
    v16bf pf = load_frag(Pw + l16 * PLD + base8);
    // O += P * V  (V^T rows are contiguous kv -> B-fragments)
#pragma unroll
    for (int t = 0; t < 8; ++t) {
      v16bf vf = load_frag(Vs + (t * 16 + l16) * VLD + kv0 + base8);
      acc[t] = wmma_bf16(pf, vf, acc[t]);
    }
  }

  // ---- epilogue: normalize and store bf16 ----
  float rinv[8];
#pragma unroll
  for (int i = 0; i < 8; ++i) rinv[i] = 1.0f / lrow[i];
#pragma unroll
  for (int t = 0; t < 8; ++t)
#pragma unroll
    for (int i = 0; i < 8; ++i) {
      int row = srow + halfr + i;
      int col = h * HEAD_DIM + t * 16 + l16;
      outg[(size_t)row * (N_HEADS * HEAD_DIM) + col] =
          (bf16)(acc[t][i] * rinv[i]);
    }
}

// ------------------------------ host launcher ------------------------------

extern "C" void kernel_launch(void* const* d_in, const int* in_sizes, int n_in,
                              void* d_out, int out_size, void* d_ws, size_t ws_size,
                              hipStream_t stream) {
  const float* x  = (const float*)d_in[0];
  const float* xk = (const float*)d_in[1];
  const float* xv = (const float*)d_in[2];
  const float* Wq = (const float*)d_in[3];
  const float* Wo = (const float*)d_in[4];
  float* out = (float*)d_out;

  char* ws = (char*)d_ws;
  size_t off = 0;
  auto take = [&](size_t bytes) -> char* {
    char* p = ws + off;
    off += (bytes + 255) & ~(size_t)255;
    return p;
  };
  bf16* xb    = (bf16*)take((size_t)S_TOT * DIM * sizeof(bf16));     // 64 MB
  bf16* wqT   = (bf16*)take((size_t)DIM * DIM * sizeof(bf16));       // 32 MB
  bf16* woT   = (bf16*)take((size_t)DIM * DIM * sizeof(bf16));       // 32 MB
  bf16* qb    = (bf16*)take((size_t)S_TOT * DIM * sizeof(bf16));     // 64 MB
  bf16* kb    = (bf16*)take((size_t)SK_TOT * N_KV_HEADS * HEAD_DIM * sizeof(bf16)); // 4 MB
  bf16* vtb   = (bf16*)take((size_t)SK_TOT * N_KV_HEADS * HEAD_DIM * sizeof(bf16)); // 4 MB
  bf16* attnb = (bf16*)take((size_t)S_TOT * DIM * sizeof(bf16));     // 64 MB
  (void)ws_size; (void)in_sizes; (void)n_in; (void)out_size;

  // 1) precision / layout prep
  {
    int n4 = S_TOT * DIM / 4;
    k_cvt_bf16<<<n4 / 256, 256, 0, stream>>>(x, xb, n4);
    dim3 g(DIM / 32, DIM / 32), blk(32, 8);
    k_transcvt<<<g, blk, 0, stream>>>(Wq, wqT, DIM, DIM);
    k_transcvt<<<g, blk, 0, stream>>>(Wo, woT, DIM, DIM);
    k_rope_k<<<(SK_TOT * N_KV_HEADS * HEAD_DIM / 2) / 256, 256, 0, stream>>>(xk, kb);
    k_vtrans<<<(SK_TOT * N_KV_HEADS * HEAD_DIM) / 256, 256, 0, stream>>>(xv, vtb);
  }
  // 2) q = x @ Wq, fused RoPE -> bf16
  {
    dim3 g(DIM / 128, S_TOT / 128);
    k_gemm_nt<<<g, 256, 0, stream>>>(xb, wqT, nullptr, qb, S_TOT, DIM, DIM, 1);
  }
  // 3) grouped cross-attention (K/V resident in 320KB WGP LDS via TDM)
  {
    size_t sh = (size_t)(K_ELE + V_ELE + P_ELE) * sizeof(bf16);  // ~276 KB
    (void)hipFuncSetAttribute((const void*)k_attn,
                              hipFuncAttributeMaxDynamicSharedMemorySize,
                              (int)sh);
    dim3 g(Q_LEN / 128, NUM_SEQS * N_HEADS);
    k_attn<<<g, 256, sh, stream>>>(qb, kb, vtb, attnb);
  }
  // 4) out = attn @ Wo, f32
  {
    dim3 g(DIM / 128, S_TOT / 128);
    k_gemm_nt<<<g, 256, 0, stream>>>(attnb, woT, out, nullptr, S_TOT, DIM, DIM, 0);
  }
}
